// Metric_Loss_64141041599004
// MI455X (gfx1250) — compile-verified, matrix operand order checked by reading, not run on hardware
//
#include <hip/hip_runtime.h>
#include <hip/hip_bf16.h>

// ---------------------------------------------------------------------------
// Metric loss for MI455X (gfx1250, wave32, WMMA).
//
//   loss = L(text) + L(mixed),  L(X): D = X X^T / 128, E = exp(1+D),
//   per pair p (rows 2p,2p+1): S = sum of E over both rows minus 2x2 block,
//   J = log(S) - D[2p,2p+1], L = 0.5*mean(J^2).
//
// Row-sums of exp(1+D) via bf16 hi/lo split WMMA (3x v_wmma_f32_16x16x32_bf16
// per 32-wide K chunk ~= f32 accuracy) spread over 6 independent accumulator
// chains (breaks the WMMA->WMMA RAW chain: deps are >=6 issue slots apart);
// B tiles staged in LDS (double buffered, bank-conflict-free padding) and
// shared by all 8 waves of a block; exact 2x2 diagonal block + D[i,j]
// recomputed in f32 in a cheap pass.
// ---------------------------------------------------------------------------

typedef __attribute__((ext_vector_type(16))) __bf16 v16bf;
typedef __attribute__((ext_vector_type(8)))  __bf16 v8bf;
typedef __attribute__((ext_vector_type(8)))  float  v8f;

#define NROWS   8192
#define EMB     128
#define NPAIRS  4096
#define INV_COS (1.0f / 128.0f)
#define MARGIN_ 1.0f

#define CHUNK_TILES 64          // column tiles per block (8 chunks cover 512)
#define SROW 136                // padded LDS row: 136 bf16 = 272 B (16B mult,
                                // +4 bank rotation per row -> conflict-free)

// ---------------------------------------------------------------- init ------
__global__ void ml_init_kernel(float* __restrict__ R, float* __restrict__ out) {
    int i = blockIdx.x * blockDim.x + threadIdx.x;
    if (i < 2 * NROWS) R[i] = 0.0f;
    if (i == 0) out[0] = 0.0f;
}

// ------------------------------------------------------- bf16 hi/lo prep ---
// Hi/Lo: [2][8192][128] bf16. Matrix 0 = text, matrix 1 = mixed
// (even rows text, odd row 2p+1 -> shape[p]).
__global__ void ml_prep_kernel(const float* __restrict__ text,
                               const float* __restrict__ shape,
                               __bf16* __restrict__ Hi,
                               __bf16* __restrict__ Lo) {
    int idx = blockIdx.x * blockDim.x + threadIdx.x;   // 0 .. 2*8192*128-1
    int k = idx & (EMB - 1);
    int r = (idx >> 7) & (NROWS - 1);
    int m = idx >> 20;                                  // 8192*128 == 2^20
    float x;
    if (m == 0)      x = text[r * EMB + k];
    else if (r & 1)  x = shape[(r >> 1) * EMB + k];
    else             x = text[r * EMB + k];
    __bf16 h = (__bf16)x;
    __bf16 l = (__bf16)(x - (float)h);
    Hi[idx] = h;
    Lo[idx] = l;
}

// ------------------------------------------------- fragment gather (A/B) ---
// 16-bit A-matrix 16x32 layout (ISA 7.12.2): lanes 0-15 hold K=[0..7]+[16..23],
// lanes 16-31 hold K=[8..15]+[24..31]: two contiguous 16B runs per lane.
// Works for global (global_load_b128) and LDS (ds_load_b128) pointers.
__device__ __forceinline__ v16bf ml_frag(const __bf16* rowp, int k0) {
    v8bf a = *(const v8bf*)(rowp + k0);
    v8bf b = *(const v8bf*)(rowp + k0 + 16);
    return __builtin_shufflevector(a, b, 0, 1, 2, 3, 4, 5, 6, 7,
                                         8, 9, 10, 11, 12, 13, 14, 15);
}

// --------------------------------------------- fused Gram + exp + reduce ---
// Block: (matrix m, column chunk of 64 tiles, row group of 8 strips).
// The 8 waves share each staged B tile; each wave owns one 16-row strip.
__global__ void __launch_bounds__(256)
ml_gram_kernel(const __bf16* __restrict__ Hi,
               const __bf16* __restrict__ Lo,
               float* __restrict__ R) {
    // [buf][hi/lo][16 rows x SROW] double-buffered B tile (17408 B total)
    __shared__ __align__(16) __bf16 sB[2][2][16 * SROW];

    const int tid  = threadIdx.x;
    const int wv   = tid >> 5;
    const int lane = tid & 31;

    const int m     = blockIdx.x >> 9;        // 512 blocks per matrix
    const int rb    = blockIdx.x & 511;
    const int chunk = rb >> 6;                // 8 column chunks
    const int rgrp  = rb & 63;                // 64 row groups of 8 strips

    const __bf16* hi = Hi + (size_t)m * NROWS * EMB;
    const __bf16* lo = Lo + (size_t)m * NROWS * EMB;

    const int mrow  = lane & 15;
    const int kbase = (lane >> 4) << 3;       // 0 or 8
    const int strip = rgrp * 8 + wv;          // this wave's 16-row strip

    // ---- A fragments (per-wave, loop invariant) ----
    const __bf16* arow_hi = hi + (size_t)(strip * 16 + mrow) * EMB;
    const __bf16* arow_lo = lo + (size_t)(strip * 16 + mrow) * EMB;
    v16bf ahi[4], alo[4];
#pragma unroll
    for (int kc = 0; kc < 4; ++kc) {
        const int k0 = kc * 32 + kbase;
        ahi[kc] = ml_frag(arow_hi, k0);
        alo[kc] = ml_frag(arow_lo, k0);
    }

    // ---- staging helpers: each B tile (16 rows x 128) is contiguous 4KB ----
    // thread copies 16B of hi and 16B of lo per tile; LDS dest is row-padded.
    const int srow   = tid >> 4;                       // 0..15
    const int schk   = tid & 15;                       // 0..15 (16B chunks)
    const int gbyte  = tid * 16;                       // global byte offset
    const int sbyte  = srow * (SROW * 2) + schk * 16;  // LDS byte offset

    const int t0 = chunk * CHUNK_TILES;

    // prologue: stage tile t0 into buffer 0
    uint4 rh = *(const uint4*)((const char*)(hi + (size_t)t0 * 16 * EMB) + gbyte);
    uint4 rl = *(const uint4*)((const char*)(lo + (size_t)t0 * 16 * EMB) + gbyte);
    *(uint4*)((char*)&sB[0][0][0] + sbyte) = rh;
    *(uint4*)((char*)&sB[0][1][0] + sbyte) = rl;

    v8f rowsum = {};
    for (int i = 0; i < CHUNK_TILES; ++i) {
        const int buf = i & 1;
        if (i + 1 < CHUNK_TILES) {   // prefetch next tile (overlaps WMMAs)
            const size_t tb = (size_t)(t0 + i + 1) * 16 * EMB;
            rh = *(const uint4*)((const char*)(hi + tb) + gbyte);
            rl = *(const uint4*)((const char*)(lo + tb) + gbyte);
        }
        __syncthreads();             // staged buf visible to all waves

        const __bf16* bh = &sB[buf][0][mrow * SROW];
        const __bf16* bl = &sB[buf][1][mrow * SROW];

        // 6 independent accumulator chains: [term(hh,hl,lh)][kc parity].
        // Dependent WMMAs (same term, kc and kc+2) are 6 issue slots apart,
        // so the matrix pipe never stalls on the RAW hazard window.
        v8f acc[6] = {};
#pragma unroll
        for (int kc = 0; kc < 4; ++kc) {
            const int k0 = kc * 32 + kbase;
            const int p  = kc & 1;
            v16bf bhi = ml_frag(bh, k0);
            v16bf blo = ml_frag(bl, k0);
            acc[0 + p] = __builtin_amdgcn_wmma_f32_16x16x32_bf16(
                false, ahi[kc], false, bhi, (short)0, acc[0 + p], false, false);
            acc[2 + p] = __builtin_amdgcn_wmma_f32_16x16x32_bf16(
                false, ahi[kc], false, blo, (short)0, acc[2 + p], false, false);
            acc[4 + p] = __builtin_amdgcn_wmma_f32_16x16x32_bf16(
                false, alo[kc], false, bhi, (short)0, acc[4 + p], false, false);
        }
        v8f d = (acc[0] + acc[1]) + (acc[2] + acc[3]) + (acc[4] + acc[5]);
#pragma unroll
        for (int v = 0; v < 8; ++v)
            rowsum[v] += __expf(MARGIN_ + d[v] * INV_COS);

        if (i + 1 < CHUNK_TILES) {   // write-back into the other buffer;
            // safe: last reads of buf^1 were in iter i-1, fenced by the
            // barrier above; next reads fenced by iter i+1's barrier.
            *(uint4*)((char*)&sB[buf ^ 1][0][0] + sbyte) = rh;
            *(uint4*)((char*)&sB[buf ^ 1][1][0] + sbyte) = rl;
        }
    }

    // C/D layout: lane holds N = lane%16, rows M = v + 8*(lane>=16).
    // Reduce over the 16 lanes sharing each row set (xor<16 stays in half).
#pragma unroll
    for (int off = 1; off < 16; off <<= 1) {
#pragma unroll
        for (int v = 0; v < 8; ++v)
            rowsum[v] += __shfl_xor((float)rowsum[v], off, 32);
    }
    if (lane == 0 || lane == 16) {
        float* Rm = R + m * NROWS + strip * 16 + ((lane >> 4) << 3);
#pragma unroll
        for (int v = 0; v < 8; ++v)
            atomicAdd(&Rm[v], (float)rowsum[v]);
    }
}

// ------------------------------------------------------------ finalize -----
__global__ void ml_finalize_kernel(const float* __restrict__ text,
                                   const float* __restrict__ shape,
                                   const float* __restrict__ R,
                                   float* __restrict__ out) {
    int tid = blockIdx.x * blockDim.x + threadIdx.x;   // 0 .. 8191
    int m = tid >> 12;
    int p = tid & (NPAIRS - 1);
    const float* xi = text + (size_t)(2 * p) * EMB;
    const float* xj = (m == 0) ? text + (size_t)(2 * p + 1) * EMB
                               : shape + (size_t)p * EMB;
    float dij = 0.f, nii = 0.f, njj = 0.f;
#pragma unroll 4
    for (int k = 0; k < EMB; k += 4) {
        float4 a = *(const float4*)(xi + k);
        float4 b = *(const float4*)(xj + k);
        dij += a.x * b.x + a.y * b.y + a.z * b.z + a.w * b.w;
        nii += a.x * a.x + a.y * a.y + a.z * a.z + a.w * a.w;
        njj += b.x * b.x + b.y * b.y + b.z * b.z + b.w * b.w;
    }
    dij *= INV_COS; nii *= INV_COS; njj *= INV_COS;
    float blk = __expf(MARGIN_ + nii) + __expf(MARGIN_ + njj)
              + 2.0f * __expf(MARGIN_ + dij);
    float S = R[m * NROWS + 2 * p] + R[m * NROWS + 2 * p + 1] - blk;
    float J = __logf(S) - dij;
    atomicAdd(out, 0.5f * J * J / (float)NPAIRS);
}

// ------------------------------------------------------------- launcher ----
extern "C" void kernel_launch(void* const* d_in, const int* in_sizes, int n_in,
                              void* d_out, int out_size, void* d_ws, size_t ws_size,
                              hipStream_t stream) {
    const float* text  = (const float*)d_in[0];   // [8192,128] f32
    const float* shape = (const float*)d_in[1];   // [4096,128] f32
    float* out = (float*)d_out;

    // ws layout: R [2][8192] f32 (64KB) | Hi [2][8192][128] bf16 (4MB) | Lo (4MB)
    float*  R  = (float*)d_ws;
    __bf16* Hi = (__bf16*)((char*)d_ws + 65536);
    __bf16* Lo = Hi + (size_t)2 * NROWS * EMB;

    ml_init_kernel<<<(2 * NROWS + 255) / 256, 256, 0, stream>>>(R, out);
    ml_prep_kernel<<<(2 * NROWS * EMB) / 256, 256, 0, stream>>>(text, shape, Hi, Lo);
    // 2 matrices * 8 column chunks * 64 row groups = 1024 blocks of 8 waves
    ml_gram_kernel<<<1024, 256, 0, stream>>>(Hi, Lo, R);
    ml_finalize_kernel<<<(2 * NPAIRS) / 256, 256, 0, stream>>>(text, shape, R, out);
}
